// OftNet_79688823210371
// MI455X (gfx1250) — compile-verified, matrix operand order checked by reading, not run on hardware
//
#include <hip/hip_runtime.h>
#include <math.h>

// ---------------- CDNA5 WMMA types ----------------
typedef __attribute__((ext_vector_type(16))) __bf16 v16bf;
typedef __attribute__((ext_vector_type(8)))  __bf16 v8bf;
typedef __attribute__((ext_vector_type(8)))  float  v8f;

#define IMC   256
#define ZC    9
#define DC    108
#define WC    124
#define NSPAT (DC * WC)     // 13392 spatial locations (== 124*108 ortho plane)
#define KOFT  (IMC * ZC)    // 2304

// Load a 16-element bf16 WMMA fragment half-pair from an LDS row of 32 K-values.
// Lane L<16: K = {0..7, 16..23}; lane>=16: K = {8..15, 24..31} (kbase = 0 or 8).
static __device__ __forceinline__ v16bf frag_ld(const __bf16* row, int kbase) {
    v8bf lo = *(const v8bf*)(row + kbase);
    v8bf hi = *(const v8bf*)(row + kbase + 16);
    return __builtin_shufflevector(lo, hi, 0,1,2,3,4,5,6,7,8,9,10,11,12,13,14,15);
}

// ---------------- f32 -> bf16 weight conversion (once per call) ----------------
__global__ void k_cvt_bf16(const float* __restrict__ in, __bf16* __restrict__ out, int n4) {
    int i = blockIdx.x * blockDim.x + threadIdx.x;
    if (i >= n4) return;
    float4 f = ((const float4*)in)[i];
    __bf16 tmp[4] = {(__bf16)f.x, (__bf16)f.y, (__bf16)f.z, (__bf16)f.w};
    *(uint2*)(out + (size_t)i * 4) = *(const uint2*)tmp;
}

// ---------------- integral image (cumsum W then H) ----------------
__global__ void k_int_row(const float* __restrict__ in, float* __restrict__ out,
                          int W, int nrows) {
    int r = blockIdx.x * blockDim.x + threadIdx.x;
    if (r >= nrows) return;
    const float* ip = in + (size_t)r * W;
    float* op = out + (size_t)r * W;
    float s = 0.f;
    for (int x = 0; x < W; ++x) { s += ip[x]; op[x] = s; }
}

__global__ void k_int_col(float* __restrict__ ii, int H, int W, int ncols) {
    int t = blockIdx.x * blockDim.x + threadIdx.x;
    if (t >= ncols) return;
    int c = t / W, x = t - c * W;
    float* p = ii + (size_t)c * H * W + x;
    float s = 0.f;
    for (int y = 0; y < H; ++y) { s += p[(size_t)y * W]; p[(size_t)y * W] = s; }
}

// ---------------- vox gather: project cell, separable 4x4 signed taps ----------------
__global__ void k_vox(const float* __restrict__ ii, const float* __restrict__ calib,
                      __bf16* __restrict__ vox, int Hf, int Wf,
                      float fx, float fy, float areaF) {
    int n = blockIdx.x * blockDim.x + threadIdx.x;
    int z = blockIdx.y;
    if (n >= NSPAT) return;
    int d = n / WC, wc = n - d * WC;

    float P[12];
#pragma unroll
    for (int i = 0; i < 12; ++i) P[i] = calib[i];

    auto proj = [&](float cx, float cy, float cz, float& nx, float& ny) {
        float h0 = P[0]*cx + P[1]*cy + P[2]*cz  + P[3];
        float h1 = P[4]*cx + P[5]*cy + P[6]*cz  + P[7];
        float h2 = P[8]*cx + P[9]*cy + P[10]*cz + P[11];
        float u = h0 / h2, v = h1 / h2;
        nx = fminf(fmaxf(u * fx - 1.f, -1.f), 1.f);
        ny = fminf(fmaxf(v * fy - 1.f, -1.f), 1.f);
    };

    float cx0 = 0.64f * d,            cx1 = 0.64f * (d + 1);
    float cy0 = 39.68f - 0.64f * wc,  cy1 = 39.68f - 0.64f * (wc + 1);
    float cz0 = 2.76f - 0.64f * z,    cz1 = 2.76f - 0.64f * (z + 1);

    float ax, ay, bx, by, cx_, cy_, dx_, dy_;
    proj(cx0, cy0, cz0, ax,  ay);    // (z,   d,   w)
    proj(cx1, cy0, cz0, bx,  by);    // (z,   d+1, w)
    proj(cx1, cy1, cz1, cx_, cy_);   // (z+1, d+1, w+1)
    proj(cx0, cy1, cz1, dx_, dy_);   // (z+1, d,   w+1)

    float bx1 = fminf(ax, bx),   by1 = fminf(ay, by);
    float bx2 = fmaxf(cx_, dx_), by2 = fmaxf(cy_, dy_);

    float area = (bx2 - bx1) * (by2 - by1) * areaF + 1e-6f;
    float s = (area > 1e-6f) ? (1.f / area) : 0.f;

    int XI[4]; float XW[4]; int YO[4]; float YW[4];
    auto xtaps = [&](float g, float sgn, int o) {
        float px = (g + 1.f) * (Wf * 0.5f) - 0.5f;
        float p0 = floorf(px); float f = px - p0; int i0 = (int)p0;
        XW[o]     = (i0     >= 0 && i0     < Wf) ? sgn * (1.f - f) : 0.f;
        XW[o + 1] = (i0 + 1 >= 0 && i0 + 1 < Wf) ? sgn * f         : 0.f;
        XI[o]     = min(max(i0, 0), Wf - 1);
        XI[o + 1] = min(max(i0 + 1, 0), Wf - 1);
    };
    auto ytaps = [&](float g, float sgn, int o) {
        float py = (g + 1.f) * (Hf * 0.5f) - 0.5f;
        float p0 = floorf(py); float f = py - p0; int i0 = (int)p0;
        YW[o]     = (i0     >= 0 && i0     < Hf) ? sgn * (1.f - f) : 0.f;
        YW[o + 1] = (i0 + 1 >= 0 && i0 + 1 < Hf) ? sgn * f         : 0.f;
        YO[o]     = min(max(i0, 0), Hf - 1) * Wf;
        YO[o + 1] = min(max(i0 + 1, 0), Hf - 1) * Wf;
    };
    // (tl + br - tr - bl) factorizes: (x2 taps - x1 taps) x (y2 taps - y1 taps)
    xtaps(bx2,  1.f, 0); xtaps(bx1, -1.f, 2);
    ytaps(by2,  1.f, 0); ytaps(by1, -1.f, 2);

    int HW = Hf * Wf;
    __bf16* vo = vox + (size_t)n * KOFT + z;
    for (int c = 0; c < IMC; ++c) {
        const float* p = ii + (size_t)c * HW;
        float acc = 0.f;
#pragma unroll
        for (int b = 0; b < 4; ++b) {
            float r = XW[0] * p[YO[b] + XI[0]] + XW[1] * p[YO[b] + XI[1]]
                    + XW[2] * p[YO[b] + XI[2]] + XW[3] * p[YO[b] + XI[3]];
            acc += YW[b] * r;
        }
        vo[c * ZC] = (__bf16)(acc * s);
    }
}

// ---------------- OFT GEMM: ortho[c][w][d] = relu(vox @ w.T + b) ----------------
// BM=64, BN=128, BK=32. 4 waves; wave -> 16 rows x 128 cols (8 WMMA accumulators).
// A and B tiles are pure copies -> CDNA5 async global->LDS DMA (ASYNCcnt).
__global__ void __launch_bounds__(128)
k_gemm_oft(const __bf16* __restrict__ vox, const __bf16* __restrict__ wgt,
           const float* __restrict__ bias, float* __restrict__ ortho, int chanBase) {
    __shared__ __bf16 At[64 * 32];
    __shared__ __bf16 Bt[128 * 32];
    int m0 = blockIdx.x * 64, n0 = blockIdx.y * 128;
    int t = threadIdx.x, lane = t & 31, wv = t >> 5;
    int r = t >> 1, half = t & 1;

    v8f zero = {0.f,0.f,0.f,0.f,0.f,0.f,0.f,0.f};
    v8f acc[8] = {zero, zero, zero, zero, zero, zero, zero, zero};

    // Clamp OOB rows to a valid row: garbage feeds only rows whose store is guarded.
    int mA = min(m0 + r, NSPAT - 1);
    const char* gA0 = (const char*)(vox + (size_t)mA * KOFT + half * 16);
    const char* gB0 = (const char*)(wgt + (size_t)(n0 + t) * KOFT);
    unsigned ldsA = (unsigned)(size_t)&At[r * 32 + half * 16];
    unsigned ldsB = (unsigned)(size_t)&Bt[t * 32];

    for (int k0 = 0; k0 < KOFT; k0 += 32) {
        const char* gA = gA0 + (size_t)k0 * 2;
        const char* gB = gB0 + (size_t)k0 * 2;
        // Async DMA: thread copies 32B of A and 64B of B straight into LDS.
        asm volatile(
            "global_load_async_to_lds_b128 %0, %2, off\n\t"
            "global_load_async_to_lds_b128 %0, %2, off offset:16\n\t"
            "global_load_async_to_lds_b128 %1, %3, off\n\t"
            "global_load_async_to_lds_b128 %1, %3, off offset:16\n\t"
            "global_load_async_to_lds_b128 %1, %3, off offset:32\n\t"
            "global_load_async_to_lds_b128 %1, %3, off offset:48\n\t"
            "s_wait_asynccnt 0x0"
            :: "v"(ldsA), "v"(ldsB), "v"(gA), "v"(gB) : "memory");
        __syncthreads();

        int kb = (lane < 16) ? 0 : 8;
        v16bf a = frag_ld(&At[(wv * 16 + (lane & 15)) * 32], kb);
#pragma unroll
        for (int j = 0; j < 8; ++j) {
            v16bf b = frag_ld(&Bt[(j * 16 + (lane & 15)) * 32], kb);
            acc[j] = __builtin_amdgcn_wmma_f32_16x16x32_bf16(
                false, a, false, b, (short)0, acc[j], false, false);
        }
        __syncthreads();
    }

    int rowBase = m0 + wv * 16 + ((lane >> 4) << 3);
    int colB = n0 + (lane & 15);
#pragma unroll
    for (int j = 0; j < 8; ++j) {
        int n = colB + j * 16;
        float bia = bias[n];
#pragma unroll
        for (int v = 0; v < 8; ++v) {
            int m = rowBase + v;
            if (m < NSPAT) {
                float val = fmaxf(acc[j][v] + bia, 0.f);
                int d = m / WC, w = m - d * WC;
                ortho[(size_t)(chanBase + n) * NSPAT + w * DC + d] = val;
            }
        }
    }
}

// ---------------- channel attention ----------------
__global__ void k_att_mean(const float* __restrict__ ortho, float* __restrict__ means) {
    __shared__ float red[256];
    int ch = blockIdx.x;
    const float* p = ortho + (size_t)ch * NSPAT;
    float s = 0.f;
    for (int i = threadIdx.x; i < NSPAT; i += 256) s += p[i];
    red[threadIdx.x] = s; __syncthreads();
    for (int o = 128; o > 0; o >>= 1) {
        if (threadIdx.x < o) red[threadIdx.x] += red[threadIdx.x + o];
        __syncthreads();
    }
    if (threadIdx.x == 0) means[ch] = red[0] * (1.f / NSPAT);
}

__global__ void k_att_scale(const float* __restrict__ means, const float* __restrict__ aw,
                            const float* __restrict__ ab, float* __restrict__ attsc) {
    int idx = blockIdx.x * blockDim.x + threadIdx.x;
    if (idx >= 5 * IMC) return;
    int s = idx >> 8, c = idx & 255;
    const float* mrow = means + (s << 8);
    const float* wrow = aw + (size_t)c * IMC;
    float acc = ab[c];
    for (int j = 0; j < IMC; ++j) acc += wrow[j] * mrow[j];
    attsc[idx] = 1.f / (1.f + expf(-acc));
}

// ---------------- implicit-GEMM 3x3 conv + BN + ReLU (WMMA) ----------------
// B-tile is a pure bf16 copy -> async DMA issued first, overlapping the im2col
// A-gather (VALU+ds_store); s_wait_asynccnt right before the barrier.
__global__ void __launch_bounds__(128)
k_conv(const float* __restrict__ in, const float* __restrict__ chanScale,
       const __bf16* __restrict__ wgt,
       const float* __restrict__ bng, const float* __restrict__ bnb,
       const float* __restrict__ bnm, const float* __restrict__ bnv,
       float* __restrict__ out,
       int IC, int H, int W, int OH, int OW, int stride) {
    __shared__ __bf16 At[64 * 32];
    __shared__ __bf16 Bt[64 * 32];
    int K = IC * 9, M = OH * OW;
    int m0 = blockIdx.x * 64, n0 = blockIdx.y * 64;
    int t = threadIdx.x, lane = t & 31, wv = t >> 5;
    int r = t >> 1, half = t & 1;
    int m = m0 + r;
    int oy = m / OW, ox = m - oy * OW;

    v8f zero = {0.f,0.f,0.f,0.f,0.f,0.f,0.f,0.f};
    v8f acc[4] = {zero, zero, zero, zero};

    const char* gB0 = (const char*)(wgt + (size_t)(n0 + r) * K + half * 16);
    unsigned ldsB = (unsigned)(size_t)&Bt[r * 32 + half * 16];

    for (int k0 = 0; k0 < K; k0 += 32) {
        // 1) kick off async B DMA (32B per thread)
        const char* gB = gB0 + (size_t)k0 * 2;
        asm volatile(
            "global_load_async_to_lds_b128 %0, %1, off\n\t"
            "global_load_async_to_lds_b128 %0, %1, off offset:16"
            :: "v"(ldsB), "v"(gB) : "memory");

        // 2) im2col A-gather with folded channel-attention scale
        __bf16* ad = &At[r * 32 + half * 16];
#pragma unroll
        for (int j = 0; j < 16; ++j) {
            int k = k0 + half * 16 + j;
            int ic = k / 9, rr = k - ic * 9;
            int ky = rr / 3, kx = rr - ky * 3;
            int iy = oy * stride - 1 + ky, ix = ox * stride - 1 + kx;
            float v = 0.f;
            if (m < M && iy >= 0 && iy < H && ix >= 0 && ix < W) {
                v = in[((size_t)ic * H + iy) * W + ix];
                if (chanScale) v *= chanScale[ic];
            }
            ad[j] = (__bf16)v;
        }

        asm volatile("s_wait_asynccnt 0x0" ::: "memory");
        __syncthreads();

        int kb = (lane < 16) ? 0 : 8;
        v16bf a = frag_ld(&At[(wv * 16 + (lane & 15)) * 32], kb);
#pragma unroll
        for (int j = 0; j < 4; ++j) {
            v16bf b = frag_ld(&Bt[(j * 16 + (lane & 15)) * 32], kb);
            acc[j] = __builtin_amdgcn_wmma_f32_16x16x32_bf16(
                false, a, false, b, (short)0, acc[j], false, false);
        }
        __syncthreads();
    }

    int rowBase = m0 + wv * 16 + ((lane >> 4) << 3);
    int colB = n0 + (lane & 15);
#pragma unroll
    for (int j = 0; j < 4; ++j) {
        int n = colB + j * 16;
        float sc = bng[n] * rsqrtf(bnv[n] + 1e-5f);
        float sh = bnb[n] - bnm[n] * sc;
#pragma unroll
        for (int v = 0; v < 8; ++v) {
            int mm = rowBase + v;
            if (mm < M) out[(size_t)n * M + mm] = fmaxf(acc[j][v] * sc + sh, 0.f);
        }
    }
}

// ---------------- host ----------------
extern "C" void kernel_launch(void* const* d_in, const int* in_sizes, int n_in,
                              void* d_out, int out_size, void* d_ws, size_t ws_size,
                              hipStream_t stream) {
    const float* feats[5];
    for (int s = 0; s < 5; ++s) feats[s] = (const float*)d_in[s];
    const float* calib   = (const float*)d_in[5];
    const float* oft_w   = (const float*)d_in[6];
    const float* oft_b   = (const float*)d_in[7];
    const float* att_w   = (const float*)d_in[8];
    const float* att_b   = (const float*)d_in[9];
    const float* conv1_w = (const float*)d_in[10];
    const float* bn1_g   = (const float*)d_in[11];
    const float* bn1_b   = (const float*)d_in[12];
    const float* bn1_m   = (const float*)d_in[13];
    const float* bn1_v   = (const float*)d_in[14];
    const float* conv2_w = (const float*)d_in[15];
    const float* bn2_g   = (const float*)d_in[16];
    const float* bn2_b   = (const float*)d_in[17];
    const float* bn2_m   = (const float*)d_in[18];
    const float* bn2_v   = (const float*)d_in[19];

    static const int   Hs[5] = {96, 48, 24, 12, 6};
    static const int   Ws[5] = {320, 160, 80, 40, 20};
    static const float SC[5] = {0.25f, 0.125f, 0.0625f, 0.03125f, 0.015625f};

    size_t iiOff[5]; size_t e = 0;
    for (int s = 0; s < 5; ++s) { iiOff[s] = e; e += (size_t)256 * Hs[s] * Ws[s]; }

    const size_t nOftW = (size_t)5 * IMC * KOFT;      // 2,949,120
    const size_t nC1W  = (size_t)IMC * 5 * IMC * 9;   // 2,949,120
    const size_t nC2W  = (size_t)IMC * IMC * 9;       //   589,824

    char* ws = (char*)d_ws;
    size_t p = 0;
    float*  ii    = (float*)(ws + p);  p += e * 4;                         p = (p + 255) & ~(size_t)255;
    __bf16* vox   = (__bf16*)(ws + p); p += (size_t)NSPAT * KOFT * 2;      p = (p + 255) & ~(size_t)255;
    float*  ortho = (float*)(ws + p);  p += (size_t)5 * IMC * NSPAT * 4;   p = (p + 255) & ~(size_t)255;
    float*  means = (float*)(ws + p);  p += 5 * IMC * 4;
    float*  attsc = (float*)(ws + p);  p += 5 * IMC * 4;                   p = (p + 255) & ~(size_t)255;
    float*  c1out = (float*)(ws + p);  p += (size_t)IMC * 62 * 54 * 4;     p = (p + 255) & ~(size_t)255;
    __bf16* wboft = (__bf16*)(ws + p); p += nOftW * 2;                     p = (p + 255) & ~(size_t)255;
    __bf16* wbc1  = (__bf16*)(ws + p); p += nC1W * 2;                      p = (p + 255) & ~(size_t)255;
    __bf16* wbc2  = (__bf16*)(ws + p); p += nC2W * 2;

    // 0) bf16 weight staging (enables async-DMA B tiles in the GEMMs)
    k_cvt_bf16<<<(int)((nOftW / 4 + 255) / 256), 256, 0, stream>>>(oft_w, wboft, (int)(nOftW / 4));
    k_cvt_bf16<<<(int)((nC1W  / 4 + 255) / 256), 256, 0, stream>>>(conv1_w, wbc1, (int)(nC1W / 4));
    k_cvt_bf16<<<(int)((nC2W  / 4 + 255) / 256), 256, 0, stream>>>(conv2_w, wbc2, (int)(nC2W / 4));

    // 1) integral images
    for (int s = 0; s < 5; ++s) {
        int nrows = 256 * Hs[s];
        k_int_row<<<(nrows + 255) / 256, 256, 0, stream>>>(feats[s], ii + iiOff[s], Ws[s], nrows);
    }
    for (int s = 0; s < 5; ++s) {
        int ncols = 256 * Ws[s];
        k_int_col<<<(ncols + 255) / 256, 256, 0, stream>>>(ii + iiOff[s], Hs[s], Ws[s], ncols);
    }

    // 2) per scale: gather vox (bf16), then WMMA GEMM -> ortho slice
    for (int s = 0; s < 5; ++s) {
        float fx = 2.f * SC[s] / Ws[s], fy = 2.f * SC[s] / Hs[s];
        float areaF = (float)(Hs[s] * Ws[s]) * 0.25f;
        k_vox<<<dim3((NSPAT + 127) / 128, ZC), 128, 0, stream>>>(
            ii + iiOff[s], calib, vox, Hs[s], Ws[s], fx, fy, areaF);
        k_gemm_oft<<<dim3((NSPAT + 63) / 64, 2), 128, 0, stream>>>(
            vox, wboft + (size_t)s * IMC * KOFT, oft_b + s * IMC, ortho, s * IMC);
    }

    // 3) channel attention scales (applied inside conv1's A-tile build)
    k_att_mean<<<5 * IMC, 256, 0, stream>>>(ortho, means);
    k_att_scale<<<(5 * IMC + 127) / 128, 128, 0, stream>>>(means, att_w, att_b, attsc);

    // 4) conv1 (s2) + BN + ReLU, conv2 (s1) + BN + ReLU -> d_out [1,256,62,54]
    k_conv<<<dim3((62 * 54 + 63) / 64, 4), 128, 0, stream>>>(
        ortho, attsc, wbc1, bn1_g, bn1_b, bn1_m, bn1_v, c1out,
        5 * IMC, WC, DC, 62, 54, 2);
    k_conv<<<dim3((62 * 54 + 63) / 64, 4), 128, 0, stream>>>(
        c1out, nullptr, wbc2, bn2_g, bn2_b, bn2_m, bn2_v, (float*)d_out,
        IMC, 62, 54, 62, 54, 1);
}
